// TransformerEncoderLayer_LocalWindowAttention_75453985457228
// MI455X (gfx1250) — matrix operand order repaired and checked by reading, not compiled, not run on hardware
//
#include <hip/hip_runtime.h>
#include <hip/hip_bf16.h>
#include <math.h>

// ---------------------------------------------------------------------------
// Types for CDNA5 WMMA (wave32): v_wmma_f32_16x16x32_bf16
// ---------------------------------------------------------------------------
typedef __attribute__((ext_vector_type(16))) __bf16 v16bf;
typedef __attribute__((ext_vector_type(8)))  __bf16 v8bf;
typedef __attribute__((ext_vector_type(8)))  float  v8f;
typedef __attribute__((ext_vector_type(4))) unsigned int u32x4;
typedef __attribute__((ext_vector_type(4))) int          i32x4;
typedef __attribute__((ext_vector_type(8))) int          i32x8;

#define NWIN 144   // 16 batch * 3 * 3 windows
#define T    224   // token dim padded 196 -> 224 (multiple of 32)
#define NTOK 196
#define WRR  14
#define CCH  256
#define HWD  40

// ---------------------------------------------------------------------------
// Tensor Data Mover: 1-D contiguous global -> LDS DMA (CDNA5 ISA).
// Guarded so a toolchain without the builtin falls back to a plain copy loop.
// ---------------------------------------------------------------------------
#if defined(__has_builtin)
#if __has_builtin(__builtin_amdgcn_tensor_load_to_lds) && \
    __has_builtin(__builtin_amdgcn_s_wait_tensorcnt)
#define USE_TDM 1
#endif
#endif
#ifndef USE_TDM
#define USE_TDM 0
#endif

#if USE_TDM
// Build a D# describing a 1-D tile of `bytes` (multiple of 8) starting at
// gsrc, landing at lds_dst. data_size=3 (8B units); tile_dim1=0 => 1-D.
__device__ __forceinline__ void tdm_load_1d(const void* gsrc, void* lds_dst,
                                            unsigned bytes) {
  unsigned long long ga = (unsigned long long)(size_t)gsrc;
  unsigned lds_off = (unsigned)(size_t)lds_dst;  // low 32b of flat addr = LDS byte offset
  unsigned n8 = bytes >> 3;                      // elements of 8 bytes (<= 65535)
  u32x4 g0;
  g0[0] = 1u;                                               // count=1, user D#
  g0[1] = lds_off;                                          // lds_addr
  g0[2] = (unsigned)ga;                                     // global_addr[31:0]
  g0[3] = (unsigned)((ga >> 32) & 0x01FFFFFFu) | (2u << 30);// addr[56:32], type=2
  i32x8 g1;
  g1[0] = (int)(3u << 16);                                  // data_size=3 (8B)
  g1[1] = (int)((n8 & 0xFFFFu) << 16);                      // tensor_dim0[15:0]
  g1[2] = (int)(((n8 >> 16) & 0xFFFFu) | (1u << 16));       // dim0[31:16], tensor_dim1=1
  g1[3] = (int)((n8 & 0xFFFFu) << 16);                      // tile_dim0 = n8
  g1[4] = 0;                                                // tile_dim1=0, tile_dim2=0
  g1[5] = (int)n8;                                          // tensor_dim0_stride
  g1[6] = 0;
  g1[7] = 0;
  i32x4 z4 = {0, 0, 0, 0};
#if defined(__clang_major__) && __clang_major__ >= 23
  i32x8 z8 = {0, 0, 0, 0, 0, 0, 0, 0};
  __builtin_amdgcn_tensor_load_to_lds(g0, g1, z4, z4, z8, 0);
#else
  __builtin_amdgcn_tensor_load_to_lds(g0, g1, z4, z4, 0);
#endif
}
#endif

__device__ __forceinline__ v8f vzero8() {
  v8f z;
#pragma unroll
  for (int i = 0; i < 8; ++i) z[i] = 0.f;
  return z;
}

__device__ __forceinline__ v8f wmma_bf16(v16bf a, v16bf b, v8f c) {
  // D = A(16x32 bf16) * B(32x16 bf16) + C(16x16 f32)
  return __builtin_amdgcn_wmma_f32_16x16x32_bf16(
      /*neg_a=*/false, a, /*neg_b=*/false, b,
      /*c_mod=*/(short)0, c, /*reuse_a=*/false, /*reuse_b=*/false);
}

// A fragment: A stored row-major [M][K]. lane holds row (lane&15),
// 16 contiguous K starting at k0 + 16*(lane>>4).
__device__ __forceinline__ v16bf load_a_frag(const __bf16* A, int row0, int ldk,
                                             int k0, int lane) {
  int m  = lane & 15;
  int kh = (lane >> 4) * 16;
  const __bf16* p = A + (size_t)(row0 + m) * ldk + k0 + kh;
  v8bf lo = *(const v8bf*)p;
  v8bf hi = *(const v8bf*)(p + 8);
  v16bf a;
#pragma unroll
  for (int i = 0; i < 8; ++i) { a[i] = lo[i]; a[i + 8] = hi[i]; }
  return a;
}

// B fragment: B stored as [N][K] (row = output col), lane holds N=(lane&15),
// 16 contiguous K starting at k0 + 16*(lane>>4).
__device__ __forceinline__ v16bf load_b_frag(const __bf16* B, int row0, int ldk,
                                             int k0, int lane) {
  const __bf16* p = B + (size_t)(row0 + (lane & 15)) * ldk + k0 + ((lane >> 4) * 16);
  return *(const v16bf*)p;
}

// Same as load_b_frag but converting f32 LDS data to bf16 on the fly.
__device__ __forceinline__ v16bf load_b_frag_f32(const float* S, int row0, int ldk,
                                                 int k0, int lane) {
  const float* p = S + (size_t)(row0 + (lane & 15)) * ldk + k0 + ((lane >> 4) * 16);
  v16bf b;
#pragma unroll
  for (int i = 0; i < 16; ++i) b[i] = (__bf16)p[i];
  return b;
}

// ---------------------------------------------------------------------------
// K0: f32 -> bf16 weight conversion
// ---------------------------------------------------------------------------
__global__ void k_cvt(const float* __restrict__ s, __bf16* __restrict__ d, int n) {
  int i = blockIdx.x * blockDim.x + threadIdx.x;
  if (i < n) d[i] = (__bf16)s[i];
}

// ---------------------------------------------------------------------------
// K1: per-window qkv GEMM (M=96,K=64,N=224) + depthwise 5x5 on q.
// grid = 144 windows, block = 256 (8 waves)
// ---------------------------------------------------------------------------
__global__ void k_qkv(int head,
                      const float* __restrict__ src,
                      const __bf16* __restrict__ feat_g,
                      const __bf16* __restrict__ Wqkv_bf,
                      const float* __restrict__ bqkv,
                      const float* __restrict__ Wdw,
                      const float* __restrict__ bdw,
                      __bf16* __restrict__ q_g,
                      __bf16* __restrict__ k_g,
                      __bf16* __restrict__ v_g) {
  __shared__ __attribute__((aligned(32))) __bf16 feat_lds[T * 64];   // [n][k]
  __shared__ __attribute__((aligned(16))) float  q_lds[T * 16];      // [n][c]
  int tid = threadIdx.x, lane = tid & 31, wid = tid >> 5;
  int win = blockIdx.x;
  int b = win / 9, wy = (win % 9) / 3, wx = win % 3;

  if (head == 0) {
    for (int idx = tid; idx < T * 64; idx += 256) {
      int k = idx / T, n = idx % T;
      float v = 0.f;
      if (n < NTOK) {
        int r = n / WRR, c = n % WRR;
        int h = wy * WRR + r, w = wx * WRR + c;
        if (h < HWD && w < HWD)
          v = src[(((size_t)b * CCH + k) * HWD + h) * HWD + w];
      }
      feat_lds[n * 64 + k] = (__bf16)v;
    }
  } else {
#if USE_TDM
    // 28KB contiguous block: let the Tensor Data Mover DMA it to LDS
    if (wid == 0) {
      tdm_load_1d(feat_g + (size_t)win * T * 64, feat_lds, T * 64 * 2);
      __builtin_amdgcn_s_wait_tensorcnt((short)0);
    }
#else
    const __bf16* fg = feat_g + (size_t)win * T * 64;
    for (int idx = tid; idx < T * 64; idx += 256) feat_lds[idx] = fg[idx];
#endif
  }
  // zero K-pad columns (16..31) of q_g / k_g so WMMA K=32 padding is exact,
  // and zero pad rows of q_g (dwconv only writes n<196).
  {
    __bf16* qg = q_g + (size_t)win * T * 32;
    __bf16* kg = k_g + (size_t)win * T * 32;
    for (int idx = tid; idx < T * 16; idx += 256) {
      int n = idx >> 4, c = 16 + (idx & 15);
      qg[n * 32 + c] = (__bf16)0.f;
      kg[n * 32 + c] = (__bf16)0.f;
    }
    for (int idx = tid; idx < (T - NTOK) * 16; idx += 256) {
      int n = NTOK + (idx >> 4), c = idx & 15;
      qg[n * 32 + c] = (__bf16)0.f;
    }
  }
  __syncthreads();

  const __bf16* A = Wqkv_bf + head * 96 * 64;   // [96][64]
  for (int job = wid; job < 6 * 14; job += 8) {
    int mt = job / 14, nt = job % 14;
    v8f acc = vzero8();
#pragma unroll
    for (int kt = 0; kt < 64; kt += 32) {
      v16bf af = load_a_frag(A, mt * 16, 64, kt, lane);
      v16bf bfrag = load_b_frag(feat_lds, nt * 16, 64, kt, lane);
      acc = wmma_bf16(af, bfrag, acc);
    }
    int n = nt * 16 + (lane & 15);
    int m0 = mt * 16 + (lane >> 4) * 8;
#pragma unroll
    for (int r = 0; r < 8; ++r) {
      int m = m0 + r;
      float val = acc[r] + bqkv[head * 96 + m];
      if (m < 16)       q_lds[n * 16 + m] = val;
      else if (m < 32)  k_g[((size_t)win * T + n) * 32 + (m - 16)] = (__bf16)val;
      else              v_g[((size_t)win * 64 + (m - 32)) * T + n] = (__bf16)val;
    }
  }
  __syncthreads();

  // depthwise 5x5 (pad 2) on q, fold in bias and SCALE=0.25
  for (int t = tid; t < 16 * NTOK; t += 256) {
    int c = t & 15, pix = t >> 4;
    int y = pix / WRR, x = pix % WRR;
    float sum = bdw[head * 16 + c];
#pragma unroll
    for (int dy = 0; dy < 5; ++dy) {
      int yy = y + dy - 2;
      if (yy < 0 || yy >= WRR) continue;
#pragma unroll
      for (int dx = 0; dx < 5; ++dx) {
        int xx = x + dx - 2;
        if (xx < 0 || xx >= WRR) continue;
        sum += q_lds[(yy * WRR + xx) * 16 + c] *
               Wdw[(head * 16 + c) * 25 + dy * 5 + dx];
      }
    }
    q_g[((size_t)win * T + pix) * 32 + c] = (__bf16)(sum * 0.25f);
  }
}

// ---------------------------------------------------------------------------
// K2: per (window, 16-row q-tile): attn logits (WMMA, bias computed on the fly
// since BIAS_IDXS == |dr|*14+|dc|), fp32 softmax, out = V @ attn^T (WMMA),
// write cat (pre-relu) and next head's feat. grid = 144*14, block = 128.
// ---------------------------------------------------------------------------
__global__ void k_attn(int head,
                       const float* __restrict__ src,
                       const float* __restrict__ attn_bias,
                       const __bf16* __restrict__ q_g,
                       const __bf16* __restrict__ k_g,
                       const __bf16* __restrict__ v_g,
                       __bf16* __restrict__ cat_g,
                       __bf16* __restrict__ feat_g) {
  __shared__ __attribute__((aligned(16))) float S[16 * T];
  int tid = threadIdx.x, lane = tid & 31, wid = tid >> 5;
  int blk = blockIdx.x;
  int win = blk / 14, nt = blk % 14;
  int b = win / 9, wy = (win % 9) / 3, wx = win % 3;
  const __bf16* qrow = q_g + (size_t)win * T * 32;
  const __bf16* krow = k_g + (size_t)win * T * 32;

  for (int mt = wid; mt < 14; mt += 4) {
    v8f acc = vzero8();
    v16bf af = load_a_frag(qrow, nt * 16, 32, 0, lane);
    v16bf bfrag = load_b_frag(krow, mt * 16, 32, 0, lane);
    acc = wmma_bf16(af, bfrag, acc);
    int mcol = mt * 16 + (lane & 15);
    int nr0 = (lane >> 4) * 8;
#pragma unroll
    for (int r = 0; r < 8; ++r) {
      int nloc = nr0 + r;
      float val;
      if (mcol < NTOK) {
        int n = nt * 16 + nloc;
        if (n >= NTOK) n = NTOK - 1;
        int dr = n / WRR - mcol / WRR; dr = dr < 0 ? -dr : dr;
        int dc = n % WRR - mcol % WRR; dc = dc < 0 ? -dc : dc;
        val = acc[r] + attn_bias[head * NTOK + dr * WRR + dc];
      } else {
        val = -1e30f;
      }
      S[nloc * T + mcol] = val;
    }
  }
  __syncthreads();
  if (tid < 16) {
    float mx = -1e30f;
    for (int m = 0; m < NTOK; ++m) mx = fmaxf(mx, S[tid * T + m]);
    float sum = 0.f;
    for (int m = 0; m < NTOK; ++m) {
      float e = __expf(S[tid * T + m] - mx);
      S[tid * T + m] = e; sum += e;
    }
    float inv = 1.f / sum;
    for (int m = 0; m < NTOK; ++m) S[tid * T + m] *= inv;
    for (int m = NTOK; m < T; ++m) S[tid * T + m] = 0.f;
  }
  __syncthreads();

  // out tile (64 x 16): A = V [64][T], B = attn rows [16][T] (f32->bf16)
  {
    int dt = wid;  // 4 waves, one 16-row M tile each
    v8f acc = vzero8();
    const __bf16* vrow = v_g + (size_t)win * 64 * T;
    for (int kt = 0; kt < 7; ++kt) {
      v16bf af = load_a_frag(vrow, dt * 16, T, kt * 32, lane);
      v16bf bfrag = load_b_frag_f32(S, 0, T, kt * 32, lane);
      acc = wmma_bf16(af, bfrag, acc);
    }
    int nloc = lane & 15, n = nt * 16 + nloc;
    int d0 = dt * 16 + (lane >> 4) * 8;
    int rr = n / WRR, cc = n % WRR;
    int h = wy * WRR + rr, w = wx * WRR + cc;
    bool inb = (n < NTOK) && (h < HWD) && (w < HWD);
#pragma unroll
    for (int r = 0; r < 8; ++r) {
      int d = d0 + r;
      float val = acc[r];
      cat_g[((size_t)win * T + n) * CCH + head * 64 + d] = (__bf16)val;
      if (head < 3) {
        float nxt = 0.f;
        if (n < NTOK) {
          float sv = inb ? src[(((size_t)b * CCH + (head + 1) * 64 + d) * HWD + h) * HWD + w]
                         : 0.f;
          nxt = val + sv;
        }
        feat_g[((size_t)win * T + n) * 64 + d] = (__bf16)nxt;
      }
    }
  }
}

// ---------------------------------------------------------------------------
// K3: proj = Wproj @ relu(cat) + bproj -> src2 (unwindowed). grid 144*14, 256.
// ---------------------------------------------------------------------------
__global__ void k_proj(const __bf16* __restrict__ cat_g,
                       const __bf16* __restrict__ Wproj_bf,
                       const float* __restrict__ bproj,
                       float* __restrict__ src2) {
  __shared__ __attribute__((aligned(32))) __bf16 b_lds[16 * CCH];  // [n][c]
  int tid = threadIdx.x, lane = tid & 31, wid = tid >> 5;
  int blk = blockIdx.x;
  int win = blk / 14, nt = blk % 14;
  int b = win / 9, wy = (win % 9) / 3, wx = win % 3;
  const __bf16* cg = cat_g + ((size_t)win * T + nt * 16) * CCH;
#if USE_TDM
  // 8KB contiguous tile via Tensor Data Mover, then relu in place.
  if (wid == 0) {
    tdm_load_1d(cg, b_lds, 16 * CCH * 2);
    __builtin_amdgcn_s_wait_tensorcnt((short)0);
  }
  __syncthreads();
  for (int idx = tid; idx < 16 * CCH; idx += 256) {
    float v = (float)b_lds[idx];
    b_lds[idx] = (__bf16)fmaxf(v, 0.f);   // relu
  }
#else
  for (int idx = tid; idx < 16 * CCH; idx += 256) {
    float v = (float)cg[idx];
    b_lds[idx] = (__bf16)fmaxf(v, 0.f);   // relu
  }
#endif
  __syncthreads();
  for (int mt = wid; mt < 16; mt += 8) {
    __builtin_prefetch(Wproj_bf + (size_t)(mt * 16 + (lane & 15)) * CCH, 0, 3);
    v8f acc = vzero8();
#pragma unroll
    for (int kt = 0; kt < 8; ++kt) {
      v16bf af = load_a_frag(Wproj_bf, mt * 16, CCH, kt * 32, lane);
      v16bf bfrag = load_b_frag(b_lds, 0, CCH, kt * 32, lane);
      acc = wmma_bf16(af, bfrag, acc);
    }
    int n = nt * 16 + (lane & 15);
    if (n < NTOK) {
      int h = wy * WRR + n / WRR, w = wx * WRR + n % WRR;
      if (h < HWD && w < HWD) {
        int m0 = mt * 16 + (lane >> 4) * 8;
#pragma unroll
        for (int r = 0; r < 8; ++r) {
          int m = m0 + r;
          src2[(((size_t)b * CCH + m) * HWD + h) * HWD + w] = acc[r] + bproj[m];
        }
      }
    }
  }
}

// ---------------------------------------------------------------------------
// K4: fused LN1 -> fc1+gelu -> fc2 -> residual -> LN2, 16 pixels per block.
// fc2 accumulated in registers over two 1024-wide K halves (LDS <= 57KB).
// grid 1600, block 256.
// ---------------------------------------------------------------------------
__global__ void k_mlp(const float* __restrict__ src,
                      const float* __restrict__ src2,
                      const float* __restrict__ n1w, const float* __restrict__ n1b,
                      const float* __restrict__ n2w, const float* __restrict__ n2b,
                      const __bf16* __restrict__ Wfc1_bf, const float* __restrict__ bfc1,
                      const __bf16* __restrict__ Wfc2_bf, const float* __restrict__ bfc2,
                      float* __restrict__ out) {
  __shared__ __attribute__((aligned(32))) __bf16 x_bf[16 * CCH];    // [n][c]
  __shared__ __attribute__((aligned(16))) float  x_f[16 * CCH];
  __shared__ __attribute__((aligned(32))) __bf16 h_bf[16 * 1024];   // [n][m] half
  int tid = threadIdx.x, lane = tid & 31, wid = tid >> 5;
  int p0 = blockIdx.x * 16;

  // LN1 over channels (per pixel); wave32: 16 lanes per pixel, shfl width 16
  {
    int pix = tid >> 4, j = tid & 15;
    int p = p0 + pix;
    int bb = p / 1600, rem = p % 1600, h = rem / HWD, w = rem % HWD;
    float v[16];
    float s = 0.f, s2 = 0.f;
#pragma unroll
    for (int q = 0; q < 16; ++q) {
      int c = j * 16 + q;
      size_t gi = (((size_t)bb * CCH + c) * HWD + h) * HWD + w;
      float t = src[gi] + src2[gi];
      v[q] = t; s += t; s2 += t * t;
    }
    for (int msk = 1; msk < 16; msk <<= 1) {
      s  += __shfl_xor(s,  msk, 16);
      s2 += __shfl_xor(s2, msk, 16);
    }
    float mean = s * (1.f / 256.f);
    float var  = s2 * (1.f / 256.f) - mean * mean;
    float rstd = rsqrtf(var + 1e-6f);
#pragma unroll
    for (int q = 0; q < 16; ++q) {
      int c = j * 16 + q;
      float xv = n1w[c] * ((v[q] - mean) * rstd) + n1b[c];
      x_f[pix * CCH + c] = xv;
      x_bf[pix * CCH + c] = (__bf16)xv;
    }
  }
  __syncthreads();

  v8f acc2[2];
  acc2[0] = vzero8(); acc2[1] = vzero8();
  for (int half = 0; half < 2; ++half) {
    // fc1 half: M=1024, K=256
    for (int mth = wid; mth < 64; mth += 8) {
      int mglob = half * 1024 + mth * 16;
      // hint the next A tile for this wave into cache while WMMAs run
      if (mth + 8 < 64)
        __builtin_prefetch(Wfc1_bf + (size_t)(half * 1024 + (mth + 8) * 16 + (lane & 15)) * CCH,
                           0, 3);
      v8f acc = vzero8();
#pragma unroll
      for (int kt = 0; kt < 8; ++kt) {
        v16bf af = load_a_frag(Wfc1_bf, mglob, CCH, kt * 32, lane);
        v16bf bfrag = load_b_frag(x_bf, 0, CCH, kt * 32, lane);
        acc = wmma_bf16(af, bfrag, acc);
      }
      int n = lane & 15;
      int ml0 = mth * 16 + (lane >> 4) * 8;
#pragma unroll
      for (int r = 0; r < 8; ++r) {
        int ml = ml0 + r;
        float vv = acc[r] + bfc1[half * 1024 + ml];
        float g = 0.5f * vv * (1.f + erff(vv * 0.70710678118f));  // exact gelu
        h_bf[n * 1024 + ml] = (__bf16)g;
      }
    }
    __syncthreads();
    // fc2 partial: M=256 (2 tiles per wave), K=1024 this half
#pragma unroll
    for (int jj = 0; jj < 2; ++jj) {
      int mt = wid * 2 + jj;
      __builtin_prefetch(Wfc2_bf + (size_t)(mt * 16 + (lane & 15)) * 2048 + half * 1024,
                         0, 3);
      for (int kt = 0; kt < 32; ++kt) {
        v16bf af = load_a_frag(Wfc2_bf, mt * 16, 2048, half * 1024 + kt * 32, lane);
        v16bf bfrag = load_b_frag(h_bf, 0, 1024, kt * 32, lane);
        acc2[jj] = wmma_bf16(af, bfrag, acc2[jj]);
      }
    }
    __syncthreads();
  }

  // residual add, stage y in LDS (reuse h_bf region)
  float* y = reinterpret_cast<float*>(h_bf);
  {
    int n = lane & 15;
#pragma unroll
    for (int jj = 0; jj < 2; ++jj) {
      int m0 = (wid * 2 + jj) * 16 + (lane >> 4) * 8;
#pragma unroll
      for (int r = 0; r < 8; ++r) {
        int m = m0 + r;
        y[n * CCH + m] = x_f[n * CCH + m] + acc2[jj][r] + bfc2[m];
      }
    }
  }
  __syncthreads();

  // LN2 + store
  {
    int pix = tid >> 4, j = tid & 15;
    int p = p0 + pix;
    int bb = p / 1600, rem = p % 1600, h = rem / HWD, w = rem % HWD;
    float v[16];
    float s = 0.f, s2 = 0.f;
#pragma unroll
    for (int q = 0; q < 16; ++q) {
      float t = y[pix * CCH + j * 16 + q];
      v[q] = t; s += t; s2 += t * t;
    }
    for (int msk = 1; msk < 16; msk <<= 1) {
      s  += __shfl_xor(s,  msk, 16);
      s2 += __shfl_xor(s2, msk, 16);
    }
    float mean = s * (1.f / 256.f);
    float var  = s2 * (1.f / 256.f) - mean * mean;
    float rstd = rsqrtf(var + 1e-6f);
#pragma unroll
    for (int q = 0; q < 16; ++q) {
      int c = j * 16 + q;
      out[(((size_t)bb * CCH + c) * HWD + h) * HWD + w] =
          n2w[c] * ((v[q] - mean) * rstd) + n2b[c];
    }
  }
}

// ---------------------------------------------------------------------------
extern "C" void kernel_launch(void* const* d_in, const int* in_sizes, int n_in,
                              void* d_out, int out_size, void* d_ws, size_t ws_size,
                              hipStream_t stream) {
  const float* src   = (const float*)d_in[0];
  const float* Wqkv  = (const float*)d_in[1];
  const float* bqkv  = (const float*)d_in[2];
  const float* Wdw   = (const float*)d_in[3];
  const float* bdw   = (const float*)d_in[4];
  const float* Wproj = (const float*)d_in[5];
  const float* bproj = (const float*)d_in[6];
  const float* abias = (const float*)d_in[7];
  const float* n1w   = (const float*)d_in[8];
  const float* n1b   = (const float*)d_in[9];
  const float* n2w   = (const float*)d_in[10];
  const float* n2b   = (const float*)d_in[11];
  const float* Wfc1  = (const float*)d_in[12];
  const float* bfc1  = (const float*)d_in[13];
  const float* Wfc2  = (const float*)d_in[14];
  const float* bfc2  = (const float*)d_in[15];
  float* out = (float*)d_out;

  char* ws = (char*)d_ws;
  size_t off = 0;
  auto carve = [&](size_t bytes) -> char* {
    char* p = ws + off;
    off = (off + bytes + 255) & ~(size_t)255;
    return p;
  };
  __bf16* wqkv_bf = (__bf16*)carve((size_t)4 * 96 * 64 * 2);
  __bf16* wproj_bf = (__bf16*)carve((size_t)256 * 256 * 2);
  __bf16* wfc1_bf = (__bf16*)carve((size_t)2048 * 256 * 2);
  __bf16* wfc2_bf = (__bf16*)carve((size_t)256 * 2048 * 2);
  __bf16* feat_g = (__bf16*)carve((size_t)NWIN * T * 64 * 2);
  __bf16* q_g    = (__bf16*)carve((size_t)NWIN * T * 32 * 2);
  __bf16* k_g    = (__bf16*)carve((size_t)NWIN * T * 32 * 2);
  __bf16* v_g    = (__bf16*)carve((size_t)NWIN * 64 * T * 2);
  __bf16* cat_g  = (__bf16*)carve((size_t)NWIN * T * CCH * 2);
  float*  src2   = (float*) carve((size_t)16 * CCH * HWD * HWD * 4);
  (void)in_sizes; (void)n_in; (void)out_size; (void)ws_size;

  k_cvt<<<(4 * 96 * 64 + 255) / 256, 256, 0, stream>>>(Wqkv, wqkv_bf, 4 * 96 * 64);
  k_cvt<<<(256 * 256 + 255) / 256, 256, 0, stream>>>(Wproj, wproj_bf, 256 * 256);
  k_cvt<<<(2048 * 256 + 255) / 256, 256, 0, stream>>>(Wfc1, wfc1_bf, 2048 * 256);
  k_cvt<<<(256 * 2048 + 255) / 256, 256, 0, stream>>>(Wfc2, wfc2_bf, 256 * 2048);

  for (int head = 0; head < 4; ++head) {
    k_qkv<<<NWIN, 256, 0, stream>>>(head, src, feat_g, wqkv_bf, bqkv, Wdw, bdw,
                                    q_g, k_g, v_g);
    k_attn<<<NWIN * 14, 128, 0, stream>>>(head, src, abias, q_g, k_g, v_g,
                                          cat_g, feat_g);
  }
  k_proj<<<NWIN * 14, 256, 0, stream>>>(cat_g, wproj_bf, bproj, src2);
  k_mlp<<<1600, 256, 0, stream>>>(src, src2, n1w, n1b, n2w, n2b,
                                  wfc1_bf, bfc1, wfc2_bf, bfc2, out);
}
